// Transformer_45483703664659
// MI455X (gfx1250) — compile-verified, matrix operand order checked by reading, not run on hardware
//
#include <hip/hip_runtime.h>
#include <cstdint>
#include <cstddef>

// ---------------------------------------------------------------------------
// Problem constants: B=64, T=512, N=256, H=4, DH=64, L=6, HID=1024.
// M = T*B = 32768 token rows; residual stream [M, N], row r = t*B + b.
// ---------------------------------------------------------------------------
#define BB   64
#define TT   512
#define NN   256
#define HH   4
#define DHD  64
#define LL   6
#define HID  1024
#define MM   (TT * BB)   // 32768

typedef unsigned short bf16_t;
typedef __attribute__((ext_vector_type(16))) __bf16          v16bf;
typedef __attribute__((ext_vector_type(16))) unsigned short  v16s;
typedef __attribute__((ext_vector_type(8)))  float           v8f;
typedef __attribute__((ext_vector_type(4)))  unsigned int    tdm_v4u;
typedef __attribute__((ext_vector_type(8)))  int             tdm_v8i;
typedef __attribute__((ext_vector_type(4)))  int             tdm_v4i;

// --------------------------- TDM availability ------------------------------
#ifndef __has_builtin
#define __has_builtin(x) 0
#endif
#if __has_builtin(__builtin_amdgcn_tensor_load_to_lds) && \
    __has_builtin(__builtin_amdgcn_s_wait_tensorcnt)
#define USE_TDM 1
#if defined(__has_include)
#if __has_include(<hip/amd_detail/amd_gfx1250_TDM.h>)
#define TDM_ARITY 6        // amdgpu-toolchain (clang-23 / therock-10.0 headers)
#else
#define TDM_ARITY 5        // ROCm 7.2 (clang-22)
#endif
#else
#define TDM_ARITY 5
#endif
#else
#define USE_TDM 0
#endif

__device__ __forceinline__ float bf2f(bf16_t h) {
  unsigned u = ((unsigned)h) << 16;
  return __builtin_bit_cast(float, u);
}
__device__ __forceinline__ bf16_t f2bf(float f) {
  unsigned u = __builtin_bit_cast(unsigned, f);
  unsigned r = (u + 0x7FFFu + ((u >> 16) & 1u)) >> 16;   // round-nearest-even
  return (bf16_t)r;
}

// D = A(16x32 bf16) x B(32x16 bf16) + C(16x16 f32) -> v_wmma_f32_16x16x32_bf16
__device__ __forceinline__ v8f wmma_bf16(const v16s& a, const v16s& b, v8f c) {
  return __builtin_amdgcn_wmma_f32_16x16x32_bf16(
      false, __builtin_bit_cast(v16bf, a),
      false, __builtin_bit_cast(v16bf, b),
      (short)0, c, false, false);
}

#if USE_TDM
// ---------------------------------------------------------------------------
// Tensor Data Mover: 2D tile load Global -> LDS (CDNA5 ISA).
// D# group0: count=1 | lds_addr | global_addr(57b) | type=2.
// D# group1: data_size=1(2B) + LDS pad config | tensor_dim0 | tensor_dim1
//            (1<<20, never OOB) | tile_dim0/1 | tensor_dim0_stride.
// cfg0 packs: data_size[17:16]=1, pad_enable[20], pad_interval[24:22],
// pad_amount[31:25].  Uniform (SGPR) operands; tracked on TENSORcnt.
// ---------------------------------------------------------------------------
__device__ __forceinline__ void tdm_load_tile(unsigned lds_off, const bf16_t* gsrc,
                                              int tile_rows, int tile_cols,
                                              int row_stride_elems, unsigned cfg0) {
  unsigned long long ga = (unsigned long long)(uintptr_t)gsrc;
  tdm_v4u g0;
  g0[0] = 1u;                                            // count=1 valid user D#
  g0[1] = lds_off;                                       // LDS byte address
  g0[2] = (unsigned)ga;                                  // global addr lo
  g0[3] = (unsigned)((ga >> 32) & 0x1FFFFFFull) | (2u << 30);  // addr hi | type=2
  tdm_v8i g1;
  g1[0] = (int)cfg0;
  g1[1] = (int)(((unsigned)row_stride_elems & 0xFFFFu) << 16); // tensor_dim0 lo16
  g1[2] = (int)(((unsigned)row_stride_elems >> 16) & 0xFFFFu); // dim0 hi16 | dim1 lo16=0
  g1[3] = (int)(16u | ((unsigned)tile_cols << 16));            // dim1 hi16 (1<<20) | tile_dim0
  g1[4] = (int)(unsigned)tile_rows;                            // tile_dim1 | tile_dim2=0
  g1[5] = (int)row_stride_elems;                               // tensor_dim0_stride lo32
  g1[6] = 0;
  g1[7] = 0;
  tdm_v4i z4 = {};
#if TDM_ARITY == 6
  tdm_v8i z8 = {};
  __builtin_amdgcn_tensor_load_to_lds(g0, g1, z4, z4, z8, 0);
#else
  __builtin_amdgcn_tensor_load_to_lds(g0, g1, z4, z4, 0);
#endif
}
#define TDM_CFG_GEMM 0xF110000u   // pad: every 32 DW (128B row) insert 8 DW (stride 80 elems)
#define TDM_CFG_ATTN 0x7110000u   // pad: every 32 DW (128B row) insert 4 DW (stride 72 elems)
#endif // USE_TDM

// ---------------------------------------------------------------------------
// fp32 -> bf16 elementwise (weight conversion)
// ---------------------------------------------------------------------------
__global__ __launch_bounds__(256) void f2bf_kernel(const float* __restrict__ src,
                                                   bf16_t* __restrict__ dst, int n) {
  int i = blockIdx.x * 256 + threadIdx.x;
  if (i < n) dst[i] = f2bf(src[i]);
}

// ---------------------------------------------------------------------------
// prep: x[t*B+b, n] = min(spikes[b,t,n], 7) * 16 + pos_emb[t, n]
// ---------------------------------------------------------------------------
__global__ __launch_bounds__(256) void prep_kernel(const float* __restrict__ spikes,
                                                   const float* __restrict__ pos,
                                                   float* __restrict__ x) {
  int i  = blockIdx.x * 256 + threadIdx.x;   // over B*T*N
  int n  = i & (NN - 1);
  int bt = i >> 8;                           // b*T + t
  int t  = bt & (TT - 1);
  int b  = bt >> 9;
  float v = fminf(spikes[i], 7.0f) * 16.0f + pos[t * NN + n];
  x[((size_t)t * BB + b) * NN + n] = v;
}

// ---------------------------------------------------------------------------
// LayerNorm over N=256, one row per wave32 (8 floats/lane), bf16 output.
// ---------------------------------------------------------------------------
__global__ __launch_bounds__(256) void ln_kernel(const float* __restrict__ x,
                                                 const float* __restrict__ s,
                                                 const float* __restrict__ b,
                                                 bf16_t* __restrict__ y) {
  int row  = blockIdx.x * 8 + (threadIdx.x >> 5);
  int lane = threadIdx.x & 31;
  const float* xr = x + (size_t)row * NN;
  float vals[8];
  float sum = 0.0f;
#pragma unroll
  for (int i = 0; i < 8; ++i) { vals[i] = xr[lane + 32 * i]; sum += vals[i]; }
#pragma unroll
  for (int off = 16; off > 0; off >>= 1) sum += __shfl_xor(sum, off, 32);
  float mean = sum * (1.0f / NN);
  float var = 0.0f;
#pragma unroll
  for (int i = 0; i < 8; ++i) { float d = vals[i] - mean; var += d * d; }
#pragma unroll
  for (int off = 16; off > 0; off >>= 1) var += __shfl_xor(var, off, 32);
  float rstd = rsqrtf(var * (1.0f / NN) + 1e-5f);
#pragma unroll
  for (int i = 0; i < 8; ++i) {
    int c = lane + 32 * i;
    y[(size_t)row * NN + c] = f2bf((vals[i] - mean) * rstd * s[c] + b[c]);
  }
}

// ---------------------------------------------------------------------------
// GEMM: out[m,n] = sum_k A[m,k] * W[n,k] + bias[n]    (W is [Nout,K] row-major)
// Block = 128 threads (4 waves).  Output tile 64 x 128; wave tile 32 x 64
// (2x4 WMMA accumulators).  K stepped by 64 through double-buffered LDS,
// staged by the Tensor Data Mover (wave 0 issues D#s; TENSORcnt pipelining),
// with LDS bank-padding done by the TDM pad engine (row stride 80 bf16).
// Epilogues: 0 bf16, 1 relu->bf16, 2 +resid->f32, 3 f32 [T,B]->[B,T].
// ---------------------------------------------------------------------------
template <int EPI>
__global__ __launch_bounds__(128) void gemm_kernel(const bf16_t* __restrict__ A,
                                                   const bf16_t* __restrict__ W,
                                                   const float* __restrict__ bias,
                                                   const float* __restrict__ resid,
                                                   bf16_t* __restrict__ outH,
                                                   float* __restrict__ outF,
                                                   int N, int K) {
  __shared__ bf16_t As[2][64][80];
  __shared__ bf16_t Bs[2][128][80];

  const int tid  = threadIdx.x;
  const int wave = tid >> 5;
  const int lane = tid & 31;
  const int half = lane >> 4;
  const int lr   = lane & 15;
  const int kh   = half * 8;

  const int row0 = blockIdx.y * 64;     // token rows
  const int col0 = blockIdx.x * 128;    // output features
  const int wr   = (wave >> 1) * 32;
  const int wc   = (wave & 1) * 64;

  v8f zero = {};
  v8f acc[2][4];
#pragma unroll
  for (int i = 0; i < 2; ++i)
#pragma unroll
    for (int j = 0; j < 4; ++j) acc[i][j] = zero;

  const int nk = K >> 6;

#if USE_TDM
  if (wave == 0) {
    tdm_load_tile((unsigned)(uintptr_t)&As[0][0][0], A + (size_t)row0 * K,
                  64, 64, K, TDM_CFG_GEMM);
    tdm_load_tile((unsigned)(uintptr_t)&Bs[0][0][0], W + (size_t)col0 * K,
                  128, 64, K, TDM_CFG_GEMM);
  }
#endif

  for (int kt = 0; kt < nk; ++kt) {
    const int bi = kt & 1;
#if USE_TDM
    if (wave == 0) {
      if (kt + 1 < nk) {
        const int k1 = (kt + 1) << 6;
        tdm_load_tile((unsigned)(uintptr_t)&As[bi ^ 1][0][0],
                      A + (size_t)row0 * K + k1, 64, 64, K, TDM_CFG_GEMM);
        tdm_load_tile((unsigned)(uintptr_t)&Bs[bi ^ 1][0][0],
                      W + (size_t)col0 * K + k1, 128, 64, K, TDM_CFG_GEMM);
        __builtin_amdgcn_s_wait_tensorcnt((short)2);  // current tiles complete
      } else {
        __builtin_amdgcn_s_wait_tensorcnt((short)0);
      }
    }
    __syncthreads();
#else
    {
      const int k1 = kt << 6;
      __syncthreads();
#pragma unroll
      for (int e = tid; e < 64 * 16; e += 128) {    // A: 64 rows x 16 chunks(4)
        int r = e >> 4, c4 = (e & 15) * 4;
        *reinterpret_cast<uint2*>(&As[bi][r][c4]) =
            *reinterpret_cast<const uint2*>(A + (size_t)(row0 + r) * K + k1 + c4);
      }
#pragma unroll
      for (int e = tid; e < 128 * 16; e += 128) {   // B: 128 rows
        int r = e >> 4, c4 = (e & 15) * 4;
        *reinterpret_cast<uint2*>(&Bs[bi][r][c4]) =
            *reinterpret_cast<const uint2*>(W + (size_t)(col0 + r) * K + k1 + c4);
      }
      if (kt + 1 < nk) {
        __builtin_prefetch(A + (size_t)(row0 + (tid & 63)) * K + k1 + 64, 0, 1);
        __builtin_prefetch(W + (size_t)(col0 + (tid & 127)) * K + k1 + 64, 0, 1);
      }
      __syncthreads();
    }
#endif
    // ---- compute on buffer bi: 2 k-chunks x (2x4) WMMAs ----
#pragma unroll
    for (int kc = 0; kc < 2; ++kc) {
      v16s af[2], bf[4];
#pragma unroll
      for (int i = 0; i < 2; ++i) {
        int r = wr + 16 * i + lr;
#pragma unroll
        for (int j = 0; j < 8; ++j) {
          af[i][j]     = As[bi][r][kc * 32 + kh + j];
          af[i][8 + j] = As[bi][r][kc * 32 + 16 + kh + j];
        }
      }
#pragma unroll
      for (int jj = 0; jj < 4; ++jj) {
        int r = wc + 16 * jj + lr;
#pragma unroll
        for (int j = 0; j < 8; ++j) {
          bf[jj][j]     = Bs[bi][r][kc * 32 + kh + j];
          bf[jj][8 + j] = Bs[bi][r][kc * 32 + 16 + kh + j];
        }
      }
#pragma unroll
      for (int i = 0; i < 2; ++i)
#pragma unroll
        for (int jj = 0; jj < 4; ++jj)
          acc[i][jj] = wmma_bf16(af[i], bf[jj], acc[i][jj]);
    }
    __syncthreads();   // everyone done reading bi before it is refilled
  }

  // ---- epilogue: C/D layout row m = v + 8*half, col n = lr ----
#pragma unroll
  for (int i = 0; i < 2; ++i)
#pragma unroll
    for (int j = 0; j < 4; ++j)
#pragma unroll
      for (int v = 0; v < 8; ++v) {
        int m = row0 + wr + 16 * i + 8 * half + v;
        int n = col0 + wc + 16 * j + lr;
        float val = acc[i][j][v] + bias[n];
        if constexpr (EPI == 0) {
          outH[(size_t)m * N + n] = f2bf(val);
        } else if constexpr (EPI == 1) {
          outH[(size_t)m * N + n] = f2bf(fmaxf(val, 0.0f));
        } else if constexpr (EPI == 2) {
          outF[(size_t)m * N + n] = resid[(size_t)m * N + n] + val;
        } else {  // decoder: row m = t*B + b  ->  out[b, t, n]
          int t = m >> 6;
          int b = m & 63;
          outF[((size_t)b * TT + t) * NN + n] = val;
        }
      }
}

// ---------------------------------------------------------------------------
// Flash-style attention.  grid = (T/64, B*H), block = 128 (4 waves).
// Q/K/V tiles staged by TDM (strided rows: stride = B*3N elems), S = Q K^T
// and O = P V via WMMA, online softmax with half-wave shuffle reductions,
// P transposed through per-wave LDS back into A-fragment layout.
// qkv: bf16 [M, 768] (q | k | v);  o: bf16 [M, 256].
// ---------------------------------------------------------------------------
__global__ __launch_bounds__(128) void attn_kernel(const bf16_t* __restrict__ qkv,
                                                   bf16_t* __restrict__ o) {
  __shared__ bf16_t Qs[64][72];
  __shared__ bf16_t Ks[64][72];
  __shared__ bf16_t Vs[64][72];
  __shared__ bf16_t Ps[4][16][72];

  const int bh   = blockIdx.y;
  const int b    = bh >> 2;       // H = 4
  const int h    = bh & 3;
  const int q0   = blockIdx.x * 64;
  const int tid  = threadIdx.x;
  const int wave = tid >> 5;
  const int lane = tid & 31;
  const int half = lane >> 4;
  const int lr   = lane & 15;
  const int kh   = half * 8;
  const float scale = 0.125f;     // 1/sqrt(DH)
  const int rowStride = BB * 3 * NN;   // 49152 elems between consecutive t

  // ---- load Q tile [64 t][64 d] ----
#if USE_TDM
  if (wave == 0) {
    tdm_load_tile((unsigned)(uintptr_t)&Qs[0][0],
                  qkv + ((size_t)q0 * BB + b) * (3 * NN) + h * DHD,
                  64, 64, rowStride, TDM_CFG_ATTN);
    __builtin_amdgcn_s_wait_tensorcnt((short)0);
  }
#else
#pragma unroll
  for (int e = tid; e < 64 * 16; e += 128) {
    int row = e >> 4;
    int c4  = (e & 15) * 4;
    *reinterpret_cast<uint2*>(&Qs[row][c4]) = *reinterpret_cast<const uint2*>(
        qkv + ((size_t)(q0 + row) * BB + b) * (3 * NN) + h * DHD + c4);
  }
#endif
  __syncthreads();

  // per-wave Q A-fragments (rows wave*16 .. +15), two 32-wide k-chunks
  v16s aq[2];
#pragma unroll
  for (int kc = 0; kc < 2; ++kc)
#pragma unroll
    for (int j = 0; j < 8; ++j) {
      aq[kc][j]     = Qs[wave * 16 + lr][kc * 32 + kh + j];
      aq[kc][8 + j] = Qs[wave * 16 + lr][kc * 32 + 16 + kh + j];
    }

  float mrow[8], lrow[8];
  v8f zero = {};
  v8f oacc[4];
#pragma unroll
  for (int j = 0; j < 4; ++j) oacc[j] = zero;
#pragma unroll
  for (int v = 0; v < 8; ++v) { mrow[v] = -3.0e38f; lrow[v] = 0.0f; }

  for (int s0 = 0; s0 < TT; s0 += 64) {
    __syncthreads();   // all waves done with Ks/Vs/Ps from previous tile
#if USE_TDM
    if (wave == 0) {
      const bf16_t* base = qkv + ((size_t)s0 * BB + b) * (3 * NN) + h * DHD;
      tdm_load_tile((unsigned)(uintptr_t)&Ks[0][0], base + NN,
                    64, 64, rowStride, TDM_CFG_ATTN);
      tdm_load_tile((unsigned)(uintptr_t)&Vs[0][0], base + 2 * NN,
                    64, 64, rowStride, TDM_CFG_ATTN);
      __builtin_amdgcn_s_wait_tensorcnt((short)0);
    }
#else
#pragma unroll
    for (int e = tid; e < 64 * 16; e += 128) {
      int row = e >> 4;
      int c4  = (e & 15) * 4;
      size_t base = ((size_t)(s0 + row) * BB + b) * (3 * NN) + h * DHD + c4;
      *reinterpret_cast<uint2*>(&Ks[row][c4]) =
          *reinterpret_cast<const uint2*>(qkv + base + NN);
      *reinterpret_cast<uint2*>(&Vs[row][c4]) =
          *reinterpret_cast<const uint2*>(qkv + base + 2 * NN);
    }
#endif
    __syncthreads();

    // ---- S = Q K^T : 16 rows x 64 keys per wave ----
    v8f sacc[4];
#pragma unroll
    for (int j = 0; j < 4; ++j) sacc[j] = zero;
#pragma unroll
    for (int j = 0; j < 4; ++j)
#pragma unroll
      for (int kc = 0; kc < 2; ++kc) {
        v16s bk;
#pragma unroll
        for (int t = 0; t < 8; ++t) {
          bk[t]     = Ks[j * 16 + lr][kc * 32 + kh + t];
          bk[8 + t] = Ks[j * 16 + lr][kc * 32 + 16 + kh + t];
        }
        sacc[j] = wmma_bf16(aq[kc], bk, sacc[j]);
      }

    // ---- online softmax (row m = v + 8*half) ----
    float mnew[8], alpha[8], rsum[8];
#pragma unroll
    for (int v = 0; v < 8; ++v) {
      float rm = -3.0e38f;
#pragma unroll
      for (int j = 0; j < 4; ++j) rm = fmaxf(rm, sacc[j][v]);
#pragma unroll
      for (int off = 1; off < 16; off <<= 1) rm = fmaxf(rm, __shfl_xor(rm, off, 32));
      mnew[v]  = fmaxf(mrow[v], rm);
      alpha[v] = __expf(scale * (mrow[v] - mnew[v]));
      mrow[v]  = mnew[v];
      rsum[v]  = 0.0f;
    }
#pragma unroll
    for (int j = 0; j < 4; ++j)
#pragma unroll
      for (int v = 0; v < 8; ++v) {
        float p = __expf(scale * (sacc[j][v] - mnew[v]));
        sacc[j][v] = p;
        rsum[v] += p;
      }
#pragma unroll
    for (int v = 0; v < 8; ++v) {
#pragma unroll
      for (int off = 1; off < 16; off <<= 1) rsum[v] += __shfl_xor(rsum[v], off, 32);
      lrow[v] = lrow[v] * alpha[v] + rsum[v];
    }
#pragma unroll
    for (int j = 0; j < 4; ++j)
#pragma unroll
      for (int v = 0; v < 8; ++v) oacc[j][v] *= alpha[v];

    // ---- P (D layout) -> LDS -> A-fragment layout ----
#pragma unroll
    for (int j = 0; j < 4; ++j)
#pragma unroll
      for (int v = 0; v < 8; ++v)
        Ps[wave][v + 8 * half][j * 16 + lr] = f2bf(sacc[j][v]);
    __syncthreads();

    v16s ap[2];
#pragma unroll
    for (int kc = 0; kc < 2; ++kc)
#pragma unroll
      for (int t = 0; t < 8; ++t) {
        ap[kc][t]     = Ps[wave][lr][kc * 32 + kh + t];
        ap[kc][8 + t] = Ps[wave][lr][kc * 32 + 16 + kh + t];
      }

    // ---- O += P V ----
#pragma unroll
    for (int j = 0; j < 4; ++j)
#pragma unroll
      for (int kc = 0; kc < 2; ++kc) {
        v16s bv;
#pragma unroll
        for (int t = 0; t < 8; ++t) {
          bv[t]     = Vs[kc * 32 + kh + t][j * 16 + lr];
          bv[8 + t] = Vs[kc * 32 + 16 + kh + t][j * 16 + lr];
        }
        oacc[j] = wmma_bf16(ap[kc], bv, oacc[j]);
      }
  }

  // ---- normalize + store ----
#pragma unroll
  for (int j = 0; j < 4; ++j)
#pragma unroll
    for (int v = 0; v < 8; ++v) {
      float val = oacc[j][v] / lrow[v];
      int t = q0 + wave * 16 + v + 8 * half;
      int d = j * 16 + lr;
      o[((size_t)t * BB + b) * NN + h * DHD + d] = f2bf(val);
    }
}

// ---------------------------------------------------------------------------
// Host-side orchestration
// ---------------------------------------------------------------------------
extern "C" void kernel_launch(void* const* d_in, const int* in_sizes, int n_in,
                              void* d_out, int out_size, void* d_ws, size_t ws_size,
                              hipStream_t stream) {
  const float* spikes = (const float*)d_in[0];
  const float* pos    = (const float*)d_in[1];
  const float* in_w   = (const float*)d_in[2];
  const float* in_b   = (const float*)d_in[3];
  const float* out_w  = (const float*)d_in[4];
  const float* out_b  = (const float*)d_in[5];
  const float* ln1_s  = (const float*)d_in[6];
  const float* ln1_b  = (const float*)d_in[7];
  const float* ln2_s  = (const float*)d_in[8];
  const float* ln2_b  = (const float*)d_in[9];
  const float* l1_w   = (const float*)d_in[10];
  const float* l1_b   = (const float*)d_in[11];
  const float* l2_w   = (const float*)d_in[12];
  const float* l2_b   = (const float*)d_in[13];
  const float* fn_s   = (const float*)d_in[14];
  const float* fn_b   = (const float*)d_in[15];
  const float* dec_w  = (const float*)d_in[16];
  const float* dec_b  = (const float*)d_in[17];

  size_t off = 0;
  auto alloc = [&](size_t bytes) -> void* {
    off = (off + 255) & ~(size_t)255;
    void* p = (char*)d_ws + off;
    off += bytes;
    return p;
  };

  float*  xf    = (float*)alloc((size_t)MM * NN * sizeof(float));
  bf16_t* ybf   = (bf16_t*)alloc((size_t)MM * NN * sizeof(bf16_t));
  bf16_t* qkvbf = (bf16_t*)alloc((size_t)MM * 3 * NN * sizeof(bf16_t));
  bf16_t* obf   = (bf16_t*)alloc((size_t)MM * NN * sizeof(bf16_t));
  bf16_t* hidbf = (bf16_t*)alloc((size_t)MM * HID * sizeof(bf16_t));
  bf16_t* inwb  = (bf16_t*)alloc((size_t)LL * 3 * NN * NN * sizeof(bf16_t));
  bf16_t* outwb = (bf16_t*)alloc((size_t)LL * NN * NN * sizeof(bf16_t));
  bf16_t* l1wb  = (bf16_t*)alloc((size_t)LL * HID * NN * sizeof(bf16_t));
  bf16_t* l2wb  = (bf16_t*)alloc((size_t)LL * NN * HID * sizeof(bf16_t));
  bf16_t* decwb = (bf16_t*)alloc((size_t)NN * NN * sizeof(bf16_t));

  auto cvt = [&](const float* s, bf16_t* d, int n) {
    f2bf_kernel<<<(n + 255) / 256, 256, 0, stream>>>(s, d, n);
  };
  cvt(in_w,  inwb,  LL * 3 * NN * NN);
  cvt(out_w, outwb, LL * NN * NN);
  cvt(l1_w,  l1wb,  LL * HID * NN);
  cvt(l2_w,  l2wb,  LL * NN * HID);
  cvt(dec_w, decwb, NN * NN);

  prep_kernel<<<(BB * TT * NN) / 256, 256, 0, stream>>>(spikes, pos, xf);

  const dim3 gemmBlk(128);
  for (int l = 0; l < LL; ++l) {
    // --- attention sub-block ---
    ln_kernel<<<MM / 8, 256, 0, stream>>>(xf, ln1_s + l * NN, ln1_b + l * NN, ybf);
    gemm_kernel<0><<<dim3(3 * NN / 128, MM / 64), gemmBlk, 0, stream>>>(
        ybf, inwb + (size_t)l * 3 * NN * NN, in_b + l * 3 * NN,
        nullptr, qkvbf, nullptr, 3 * NN, NN);
    attn_kernel<<<dim3(TT / 64, BB * HH), gemmBlk, 0, stream>>>(qkvbf, obf);
    gemm_kernel<2><<<dim3(NN / 128, MM / 64), gemmBlk, 0, stream>>>(
        obf, outwb + (size_t)l * NN * NN, out_b + l * NN,
        xf, nullptr, xf, NN, NN);
    // --- FFN sub-block ---
    ln_kernel<<<MM / 8, 256, 0, stream>>>(xf, ln2_s + l * NN, ln2_b + l * NN, ybf);
    gemm_kernel<1><<<dim3(HID / 128, MM / 64), gemmBlk, 0, stream>>>(
        ybf, l1wb + (size_t)l * HID * NN, l1_b + l * HID,
        nullptr, hidbf, nullptr, HID, NN);
    gemm_kernel<2><<<dim3(NN / 128, MM / 64), gemmBlk, 0, stream>>>(
        hidbf, l2wb + (size_t)l * NN * HID, l2_b + l * NN,
        xf, nullptr, xf, NN, HID);
  }

  // final LN + decoder (fused [T,B]->[B,T] transpose into epilogue)
  ln_kernel<<<MM / 8, 256, 0, stream>>>(xf, fn_s, fn_b, ybf);
  gemm_kernel<3><<<dim3(NN / 128, MM / 64), gemmBlk, 0, stream>>>(
      ybf, decwb, dec_b, nullptr, nullptr, (float*)d_out, NN, NN);
}
